// Model_77077483094358
// MI455X (gfx1250) — matrix operand order repaired and checked by reading, not compiled, not run on hardware
//
#include <hip/hip_runtime.h>
#include <math.h>

// ---------------- problem constants (from reference) ----------------
#define NN   2000   // codes
#define KP   2016   // NN padded to multiple of 32 (K dim of big GEMMs)
#define EE   32     // visits
#define DD   128    // embed dim
#define HDD  256    // hidden
#define OO   1000   // classes
#define BB   16     // batch
#define CAP  1024   // max codes per visit we record (expected ~40)
#define EPSF 1e-12f

typedef __attribute__((ext_vector_type(16))) __bf16        v16bf;
typedef __attribute__((ext_vector_type(8)))  float         v8f;
typedef __attribute__((ext_vector_type(4)))  unsigned int  v4u;

__device__ __forceinline__ __bf16 f2bf(float f) { return (__bf16)f; }

union Frag { v16bf v; v4u u[2]; };

// ======================================================================
// Core GEMM: C[M x Ncol] = A[M x Kp] * B[Kp x Ncol] (+bias, +relu).
//   A  : row-major [M x Kp], bf16, K fully padded (zeros) to multiple of 32
//   Bt : B TRANSPOSED, row-major [Ncol x Kp], bf16, zero-padded
// One wave computes a 32x32 tile = 4 accumulators, 4 WMMAs per K-step.
// All fragment loads are contiguous 128-bit loads; no guards in K-loop,
// so EXEC is all-ones through every v_wmma (ISA requirement).
// ======================================================================
__global__ void k_gemm_bf16(const __bf16* __restrict__ A,
                            const __bf16* __restrict__ Bt,
                            const float* __restrict__ bias,
                            float* __restrict__ C,
                            int M, int Ncol, int Kp, int relu)
{
    const int tilesN = (Ncol + 31) >> 5;
    const int tilesM = (M + 31) >> 5;
    const int wave   = blockIdx.x * (blockDim.x >> 5) + (threadIdx.x >> 5);
    const int tM     = wave / tilesN;
    const int tN     = wave - tM * tilesN;
    if (tM >= tilesM) return;

    const int lane = threadIdx.x & 31;
    const int half = lane >> 4;          // K-half selector per ISA layout
    const int r    = lane & 15;

    // load rows/cols, clamped to valid memory (garbage only feeds tiles
    // whose stores are masked out below -> EXEC stays all-ones)
    const int m0 = tM * 32 + r,  m1 = m0 + 16;
    const int n0 = tN * 32 + r,  n1 = n0 + 16;
    const int ma0 = (m0 < M)    ? m0 : 0;
    const int ma1 = (m1 < M)    ? m1 : 0;
    const int na0 = (n0 < Ncol) ? n0 : 0;
    const int na1 = (n1 < Ncol) ? n1 : 0;

    // A frag: elems 0..7 <-> K = k0+half*8+e ; elems 8..15 <-> K = k0+16+half*8+e
    const __bf16* pa0 = A  + (size_t)ma0 * Kp + half * 8;
    const __bf16* pa1 = A  + (size_t)ma1 * Kp + half * 8;
    // B frag: elems 0..7 <-> K = k0+half*8+e ; elems 8..15 <-> K = k0+16+half*8+e of column n
    const __bf16* pb0 = Bt + (size_t)na0 * Kp + half * 8;
    const __bf16* pb1 = Bt + (size_t)na1 * Kp + half * 8;

    v8f acc00 = {}, acc01 = {}, acc10 = {}, acc11 = {};
    for (int k0 = 0; k0 < Kp; k0 += 32) {
        Frag a0, a1, b0, b1;
        a0.u[0] = *(const v4u*)pa0;       a0.u[1] = *(const v4u*)(pa0 + 16);
        a1.u[0] = *(const v4u*)pa1;       a1.u[1] = *(const v4u*)(pa1 + 16);
        b0.u[0] = *(const v4u*)pb0;       b0.u[1] = *(const v4u*)(pb0 + 16);
        b1.u[0] = *(const v4u*)pb1;       b1.u[1] = *(const v4u*)(pb1 + 16);
        acc00 = __builtin_amdgcn_wmma_f32_16x16x32_bf16(false, a0.v, false, b0.v, (short)0, acc00, false, false);
        acc01 = __builtin_amdgcn_wmma_f32_16x16x32_bf16(false, a0.v, false, b1.v, (short)0, acc01, false, false);
        acc10 = __builtin_amdgcn_wmma_f32_16x16x32_bf16(false, a1.v, false, b0.v, (short)0, acc10, false, false);
        acc11 = __builtin_amdgcn_wmma_f32_16x16x32_bf16(false, a1.v, false, b1.v, (short)0, acc11, false, false);
        pa0 += 32; pa1 += 32; pb0 += 32; pb1 += 32;
    }

    // C layout: VGPR v -> row (v + 8*half), col (lane&15) within each 16x16 tile
    const int cn0 = tN * 32 + r, cn1 = cn0 + 16;
#pragma unroll
    for (int v = 0; v < 8; ++v) {
        const int cm0 = tM * 32 + v + 8 * half;
        const int cm1 = cm0 + 16;
        float b0v = 0.f, b1v = 0.f;
        if (bias) { b0v = (cn0 < Ncol) ? bias[cn0] : 0.f; b1v = (cn1 < Ncol) ? bias[cn1] : 0.f; }
        float x00 = acc00[v] + b0v, x01 = acc01[v] + b1v;
        float x10 = acc10[v] + b0v, x11 = acc11[v] + b1v;
        if (relu) { x00 = fmaxf(x00, 0.f); x01 = fmaxf(x01, 0.f);
                    x10 = fmaxf(x10, 0.f); x11 = fmaxf(x11, 0.f); }
        if (cm0 < M && cn0 < Ncol) C[(size_t)cm0 * Ncol + cn0] = x00;
        if (cm0 < M && cn1 < Ncol) C[(size_t)cm0 * Ncol + cn1] = x01;
        if (cm1 < M && cn0 < Ncol) C[(size_t)cm1 * Ncol + cn0] = x10;
        if (cm1 < M && cn1 < Ncol) C[(size_t)cm1 * Ncol + cn1] = x11;
    }
}

// ---------------- conversions ----------------
// plain contiguous f32 -> bf16
__global__ void k_cvt(const float* __restrict__ s, __bf16* __restrict__ d, size_t n)
{
    size_t i = (size_t)blockIdx.x * blockDim.x + threadIdx.x;
    if (i < n) d[i] = f2bf(s[i]);
}

// f32 [rows x cols] -> bf16 with transpose: d[c*ldd + r] = s[r*cols + c]
__global__ void k_cvt_T(const float* __restrict__ s, __bf16* __restrict__ d,
                        int rows, int cols, int ldd)
{
    size_t i = (size_t)blockIdx.x * blockDim.x + threadIdx.x;
    if (i < (size_t)rows * cols) {
        int rr = (int)(i / cols), cc = (int)(i - (size_t)rr * cols);
        d[(size_t)cc * ldd + rr] = f2bf(s[i]);
    }
}

// f32 [NN x NN] -> bf16 [NN x KP] (row-padded)
__global__ void k_cvt_pad(const float* __restrict__ s, __bf16* __restrict__ d)
{
    size_t i = (size_t)blockIdx.x * blockDim.x + threadIdx.x;
    if (i < (size_t)NN * NN) {
        int rr = (int)(i / NN), cc = (int)(i - (size_t)rr * NN);
        d[(size_t)rr * KP + cc] = f2bf(s[i]);
    }
}

// ---------------- layernorm over D (block = D threads, block per row) ----------------
__global__ void k_layernorm(const float* __restrict__ x, const float* __restrict__ g,
                            const float* __restrict__ b, float* __restrict__ y)
{
    const int row = blockIdx.x, d = threadIdx.x;
    __shared__ float red[DD];
    float v = x[(size_t)row * DD + d];
    red[d] = v; __syncthreads();
    for (int s = DD / 2; s > 0; s >>= 1) { if (d < s) red[d] += red[d + s]; __syncthreads(); }
    float mean = red[0] / DD; __syncthreads();
    float diff = v - mean;
    red[d] = diff * diff; __syncthreads();
    for (int s = DD / 2; s > 0; s >>= 1) { if (d < s) red[d] += red[d + s]; __syncthreads(); }
    float var = red[0] / DD;
    y[(size_t)row * DD + d] = diff * rsqrtf(var + 1e-5f) * g[d] + b[d];
}

// ---------------- per-patient graph construction ----------------
// serial scan per visit -> deterministic code ordering
__global__ void k_build_codes(const float* __restrict__ cx, const int* __restrict__ lens,
                              int p, int* __restrict__ cnt, int* __restrict__ codes,
                              float* __restrict__ De)
{
    int e = blockIdx.x;
    if (threadIdx.x != 0) return;
    int len = lens[p], c = 0;
    if (e < len) {
        for (int i = 0; i < NN; ++i)
            if (cx[(size_t)e * NN + i] > 0.5f) { if (c < CAP) codes[e * CAP + c] = i; ++c; }
    }
    if (c > CAP) c = CAP;
    cnt[e] = c;
    De[e] = (float)c;
}

__global__ void k_dv(const float* __restrict__ cx, const int* __restrict__ lens,
                     int p, float* __restrict__ Dv)
{
    int i = blockIdx.x * blockDim.x + threadIdx.x;
    if (i >= NN) return;
    int len = lens[p];
    float s = 0.f;
    for (int e = 0; e < EE; ++e)
        if (e < len) s += cx[(size_t)e * NN + i];
    Dv[i] = s;
}

__global__ void k_denom(const float* __restrict__ Dv, float* __restrict__ denom)
{
    __shared__ float red[256];
    float s = 0.f;
    for (int i = threadIdx.x; i < NN; i += 256) s += (Dv[i] > 0.f) ? 1.f : 0.f;
    red[threadIdx.x] = s; __syncthreads();
    for (int st = 128; st > 0; st >>= 1) { if (threadIdx.x < st) red[threadIdx.x] += red[threadIdx.x + st]; __syncthreads(); }
    if (threadIdx.x == 0) denom[0] = fmaxf(red[0], 1.0f);
}

__global__ void k_transitions(const int* __restrict__ lens, int p,
                              const int* __restrict__ cnt, const int* __restrict__ codes,
                              float* __restrict__ A)
{
    int t = blockIdx.x;
    int len = lens[p];
    if (t + 1 >= len) return;
    int c0 = cnt[t], c1 = cnt[t + 1];
    int total = c0 * c1;
    for (int idx = threadIdx.x; idx < total; idx += blockDim.x) {
        int i = codes[t * CAP + idx / c1];
        int j = codes[(t + 1) * CAP + idx % c1];
        if (i != j) A[(size_t)i * NN + j] = 1.0f;   // races write same value: fine
    }
}

__global__ void k_rowsum(const float* __restrict__ A, float* __restrict__ rs)
{
    int i = blockIdx.x;
    __shared__ float red[256];
    float s = 0.f;
    for (int j = threadIdx.x; j < NN; j += 256) s += A[(size_t)i * NN + j];
    red[threadIdx.x] = s; __syncthreads();
    for (int st = 128; st > 0; st >>= 1) { if (threadIdx.x < st) red[threadIdx.x] += red[threadIdx.x + st]; __syncthreads(); }
    if (threadIdx.x == 0) rs[i] = red[0];
}

__global__ void k_colsum(const float* __restrict__ A, float* __restrict__ cs)
{
    int j = blockIdx.x * blockDim.x + threadIdx.x;
    if (j >= NN) return;
    float s = 0.f;
    for (int i = 0; i < NN; ++i) s += A[(size_t)i * NN + j];
    cs[j] = s;
}

// normalized A -> bf16 [NN x KP] and bf16 A^T [NN x KP] in one pass
__global__ void k_norm_A(const float* __restrict__ A, const float* __restrict__ rs,
                         const float* __restrict__ cs, __bf16* __restrict__ Ab,
                         __bf16* __restrict__ ATb)
{
    size_t idx = (size_t)blockIdx.x * blockDim.x + threadIdx.x;
    if (idx >= (size_t)NN * NN) return;
    int i = (int)(idx / NN), j = (int)(idx - (size_t)i * NN);
    float v = A[idx];
    if (v != 0.0f)
        v = v * rsqrtf(fmaxf(rs[i], EPSF)) * rsqrtf(fmaxf(cs[j], EPSF));
    __bf16 bv = f2bf(v);
    Ab[(size_t)i * KP + j] = bv;
    ATb[(size_t)j * KP + i] = bv;
}

// ---------------- hypergraph conv (Hf is N x E, E=32 -> VALU kernels) ----------------
__global__ void k_hg_t(const float* __restrict__ X, const int* __restrict__ cnt,
                       const int* __restrict__ codes, const float* __restrict__ Dv,
                       const float* __restrict__ De, const int* __restrict__ lens,
                       int p, float* __restrict__ T)
{
    int e = blockIdx.x, d = threadIdx.x;   // blockDim = DD
    int len = lens[p];
    float s = 0.f;
    if (e < len) {
        int c = cnt[e];
        for (int q = 0; q < c; ++q) {
            int i = codes[e * CAP + q];
            s += X[(size_t)i * DD + d] * rsqrtf(fmaxf(Dv[i], EPSF));
        }
        s *= 1.0f / fmaxf(De[e], EPSF);
    }
    T[e * DD + d] = s;
}

__global__ void k_hg_out(const float* __restrict__ cx, const int* __restrict__ lens,
                         int p, const float* __restrict__ T, const float* __restrict__ Dv,
                         float* __restrict__ out)
{
    int i = blockIdx.x;
    __shared__ float flag[EE];
    int len = lens[p];
    if (threadIdx.x < EE) {
        int e = threadIdx.x;
        flag[e] = (e < len && cx[(size_t)e * NN + i] > 0.5f) ? 1.f : 0.f;
    }
    __syncthreads();
    int d = threadIdx.x;   // blockDim = DD
    float s = 0.f;
    for (int e = 0; e < EE; ++e) s += flag[e] * T[e * DD + d];
    out[(size_t)i * DD + d] = s * rsqrtf(fmaxf(Dv[i], EPSF));
}

// ---------------- masked mean pool over N rows -> HD vector ----------------
__global__ void k_pool(const float* __restrict__ M, const float* __restrict__ Dv,
                       const float* __restrict__ denom, float* __restrict__ evec)
{
    int h = threadIdx.x;   // blockDim = HDD
    float s = 0.f;
    for (int i = 0; i < NN; ++i)
        if (Dv[i] > 0.f) s += M[(size_t)i * HDD + h];
    evec[h] = s / denom[0];
}

// ---------------- gates + residual fuse ----------------
__global__ void k_gate(const float* __restrict__ eh, const float* __restrict__ ec,
                       const float* __restrict__ ei,
                       const float* __restrict__ gwh, const float* __restrict__ gbh,
                       const float* __restrict__ gwc, const float* __restrict__ gbc,
                       const float* __restrict__ gwi, const float* __restrict__ gbi,
                       float* __restrict__ res)
{
    __shared__ float r0[HDD], r1[HDD], r2[HDD];
    int h = threadIdx.x;
    float vh = eh[h], vc = ec[h], vi = ei[h];
    r0[h] = vh * gwh[h]; r1[h] = vc * gwc[h]; r2[h] = vi * gwi[h];
    __syncthreads();
    for (int s = HDD / 2; s > 0; s >>= 1) {
        if (h < s) { r0[h] += r0[h + s]; r1[h] += r1[h + s]; r2[h] += r2[h + s]; }
        __syncthreads();
    }
    float gh = 1.f / (1.f + expf(-(r0[0] + gbh[0])));
    float gc = 1.f / (1.f + expf(-(r1[0] + gbc[0])));
    float gi = 1.f / (1.f + expf(-(r2[0] + gbi[0])));
    res[h] = gh * vh + gc * vc + gi * vi;
}

// ---------------- InfoNCE(E1,E2) + InfoNCE(E1,E3) ----------------
__global__ void k_loss(const float* __restrict__ E1, const float* __restrict__ E2,
                       const float* __restrict__ E3, float* __restrict__ lossOut)
{
    __shared__ float a[BB * HDD], b2[BB * HDD], b3[BB * HDD];
    __shared__ float n1[BB], n2[BB], n3[BB];
    __shared__ float s12[BB * BB], s13[BB * BB];
    int t = threadIdx.x;   // 256 threads
    for (int idx = t; idx < BB * HDD; idx += 256) { a[idx] = E1[idx]; b2[idx] = E2[idx]; b3[idx] = E3[idx]; }
    __syncthreads();
    if (t < BB) {
        float s = 0.f, u = 0.f, w = 0.f;
        for (int h = 0; h < HDD; ++h) {
            s += a[t * HDD + h] * a[t * HDD + h];
            u += b2[t * HDD + h] * b2[t * HDD + h];
            w += b3[t * HDD + h] * b3[t * HDD + h];
        }
        n1[t] = fmaxf(sqrtf(s), EPSF);
        n2[t] = fmaxf(sqrtf(u), EPSF);
        n3[t] = fmaxf(sqrtf(w), EPSF);
    }
    __syncthreads();
    {
        int i = t >> 4, j = t & 15;   // 256 threads = 16x16 pairs
        float d12 = 0.f, d13 = 0.f;
        for (int h = 0; h < HDD; ++h) {
            float av = a[i * HDD + h] / n1[i];
            d12 += av * b2[j * HDD + h] / n2[j];
            d13 += av * b3[j * HDD + h] / n3[j];
        }
        s12[i * BB + j] = d12;
        s13[i * BB + j] = d13;
    }
    __syncthreads();
    if (t == 0) {
        const float tau = 0.4f;
        float total = 0.f;
        for (int pass = 0; pass < 2; ++pass) {
            const float* sm = pass ? s13 : s12;
            float l = 0.f;
            for (int k = 0; k < BB; ++k) {
                float pos = expf(sm[k * BB + k] / tau);
                float neg = 0.f;
                for (int j = 0; j < BB; ++j) neg += expf(sm[k * BB + j] / tau);
                l += -logf(pos / (neg + 1e-8f) + 1e-8f);
            }
            total += l / BB;
        }
        lossOut[0] = total;
    }
}

// ======================================================================
// host-side launch helpers
// ======================================================================
static inline void gemm(const __bf16* A, const __bf16* Bt, const float* bias, float* C,
                        int M, int Ncol, int Kp, int relu, hipStream_t s)
{
    int tiles  = ((M + 31) / 32) * ((Ncol + 31) / 32);
    int blocks = (tiles + 7) / 8;                 // 8 waves (wave32) per 256-thread block
    k_gemm_bf16<<<blocks, 256, 0, s>>>(A, Bt, bias, C, M, Ncol, Kp, relu);
}

static inline void cvt(const float* s, __bf16* d, size_t n, hipStream_t st)
{
    k_cvt<<<(unsigned)((n + 255) / 256), 256, 0, st>>>(s, d, n);
}

static inline void cvtT(const float* s, __bf16* d, int rows, int cols, int ldd, hipStream_t st)
{
    k_cvt_T<<<(unsigned)(((size_t)rows * cols + 255) / 256), 256, 0, st>>>(s, d, rows, cols, ldd);
}

extern "C" void kernel_launch(void* const* d_in, const int* in_sizes, int n_in,
                              void* d_out, int out_size, void* d_ws, size_t ws_size,
                              hipStream_t stream)
{
    (void)in_sizes; (void)n_in; (void)out_size; (void)ws_size;
    const float* code_x     = (const float*)d_in[0];
    const int*   lens       = (const int*)  d_in[1];
    const float* embeddings = (const float*)d_in[2];
    const float* adj        = (const float*)d_in[3];
    const float* ln_g       = (const float*)d_in[4];
    const float* ln_b       = (const float*)d_in[5];
    const float* W_h        = (const float*)d_in[6];
    const float* b_h        = (const float*)d_in[7];
    const float* W_c        = (const float*)d_in[8];
    const float* b_c        = (const float*)d_in[9];
    const float* W_i        = (const float*)d_in[10];
    const float* b_i        = (const float*)d_in[11];
    const float* gw_h       = (const float*)d_in[12];
    const float* gb_h       = (const float*)d_in[13];
    const float* gw_c       = (const float*)d_in[14];
    const float* gb_c       = (const float*)d_in[15];
    const float* gw_i       = (const float*)d_in[16];
    const float* gb_i       = (const float*)d_in[17];
    const float* W_cls      = (const float*)d_in[18];
    const float* b_cls      = (const float*)d_in[19];
    float* out = (float*)d_out;

    // ---- bump allocator over d_ws ----
    size_t off = 0;
    auto alloc = [&](size_t bytes) -> void* {
        void* p = (char*)d_ws + off;
        off += (bytes + 255) & ~(size_t)255;
        return p;
    };
    __bf16* adjB   = (__bf16*)alloc((size_t)NN * KP * 2);   // adj,   A-role [NN x KP]
    __bf16* adjTB  = (__bf16*)alloc((size_t)NN * KP * 2);   // adj^T, A-role
    __bf16* Ab     = (__bf16*)alloc((size_t)NN * KP * 2);   // patient A
    __bf16* ATb    = (__bf16*)alloc((size_t)NN * KP * 2);   // patient A^T
    __bf16* embBt  = (__bf16*)alloc((size_t)DD * KP * 2);   // emb^T, B-role [DD x KP]
    __bf16* tBt    = (__bf16*)alloc((size_t)DD * KP * 2);   // temp  B-role [DD x KP]
    __bf16* tA     = (__bf16*)alloc((size_t)NN * DD * 2);   // temp  A-role [NN x DD]
    __bf16* WhBt   = (__bf16*)alloc((size_t)HDD * DD * 2);  // W_h^T  [HDD x DD]
    __bf16* WcBt   = (__bf16*)alloc((size_t)HDD * DD * 2);
    __bf16* WiBt   = (__bf16*)alloc((size_t)HDD * DD * 2);
    __bf16* WclsBt = (__bf16*)alloc((size_t)OO * HDD * 2);  // W_cls^T [OO x HDD]
    __bf16* resB   = (__bf16*)alloc((size_t)BB * HDD * 2);
    float*  xbuf   = (float*)alloc((size_t)NN * DD * 4);    // icd_emb working/result
    float*  ybuf   = (float*)alloc((size_t)NN * DD * 4);
    float*  zbuf   = (float*)alloc((size_t)NN * DD * 4);
    float*  A      = (float*)alloc((size_t)NN * NN * 4);
    float*  rs     = (float*)alloc((size_t)NN * 4);
    float*  cs     = (float*)alloc((size_t)NN * 4);
    float*  Dv     = (float*)alloc((size_t)NN * 4);
    float*  De     = (float*)alloc((size_t)EE * 4);
    int*    cnt    = (int*)alloc((size_t)EE * 4);
    int*    codes  = (int*)alloc((size_t)EE * CAP * 4);
    float*  T      = (float*)alloc((size_t)EE * DD * 4);
    float*  outB   = (float*)alloc((size_t)NN * HDD * 4);
    float*  E1     = (float*)alloc((size_t)BB * HDD * 4);
    float*  E2     = (float*)alloc((size_t)BB * HDD * 4);
    float*  E3     = (float*)alloc((size_t)BB * HDD * 4);
    float*  resf   = (float*)alloc((size_t)BB * HDD * 4);
    float*  denom  = (float*)alloc(256);

    // ---- zero pad regions (and K-padded buffers) once per call ----
    hipMemsetAsync(adjB,  0, (size_t)NN * KP * 2, stream);
    hipMemsetAsync(adjTB, 0, (size_t)NN * KP * 2, stream);
    hipMemsetAsync(Ab,    0, (size_t)NN * KP * 2, stream);
    hipMemsetAsync(ATb,   0, (size_t)NN * KP * 2, stream);
    hipMemsetAsync(embBt, 0, (size_t)DD * KP * 2, stream);
    hipMemsetAsync(tBt,   0, (size_t)DD * KP * 2, stream);

    // ---- one-time weight / adjacency conversions ----
    k_cvt_pad<<<(unsigned)(((size_t)NN * NN + 255) / 256), 256, 0, stream>>>(adj, adjB);
    cvtT(adj, adjTB, NN, NN, KP, stream);            // adj^T with K padding
    cvtT(embeddings, embBt, NN, DD, KP, stream);     // [DD x KP] (= emb^T, B-role)
    cvtT(W_h, WhBt, DD, HDD, DD, stream);            // [HDD x DD]
    cvtT(W_c, WcBt, DD, HDD, DD, stream);
    cvtT(W_i, WiBt, DD, HDD, DD, stream);
    cvtT(W_cls, WclsBt, HDD, OO, HDD, stream);       // [OO x HDD]

    // ---- icd_conv: 3x (adj.T @ (adj @ x)) + layernorm ----
    for (int it = 0; it < 3; ++it) {
        const float* src = (it == 0) ? embeddings : xbuf;
        cvtT(src, tBt, NN, DD, KP, stream);          // x^T as B-role
        gemm(adjB, tBt, nullptr, ybuf, NN, DD, KP, 0, stream);
        cvtT(ybuf, tBt, NN, DD, KP, stream);
        gemm(adjTB, tBt, nullptr, zbuf, NN, DD, KP, 0, stream);
        k_layernorm<<<NN, DD, 0, stream>>>(zbuf, ln_g, ln_b, xbuf);
    }
    // xbuf == icd_emb

    // ---- per-patient pipeline ----
    for (int p = 0; p < BB; ++p) {
        const float* cx = code_x + (size_t)p * EE * NN;

        hipMemsetAsync(A, 0, (size_t)NN * NN * sizeof(float), stream);
        k_build_codes<<<EE, 32, 0, stream>>>(cx, lens, p, cnt, codes, De);
        k_dv<<<(NN + 255) / 256, 256, 0, stream>>>(cx, lens, p, Dv);
        k_denom<<<1, 256, 0, stream>>>(Dv, denom);
        k_transitions<<<EE - 1, 256, 0, stream>>>(lens, p, cnt, codes, A);
        k_rowsum<<<NN, 256, 0, stream>>>(A, rs);
        k_colsum<<<(NN + 255) / 256, 256, 0, stream>>>(A, cs);
        k_norm_A<<<(unsigned)(((size_t)NN * NN + 255) / 256), 256, 0, stream>>>(A, rs, cs, Ab, ATb);

        // causal branch: relu((A.T @ (A @ emb)) @ W_c + b_c), pooled -> E2
        gemm(Ab, embBt, nullptr, ybuf, NN, DD, KP, 0, stream);
        cvtT(ybuf, tBt, NN, DD, KP, stream);
        gemm(ATb, tBt, nullptr, zbuf, NN, DD, KP, 0, stream);
        cvt(zbuf, tA, (size_t)NN * DD, stream);
        gemm(tA, WcBt, b_c, outB, NN, HDD, DD, 1, stream);
        k_pool<<<1, HDD, 0, stream>>>(outB, Dv, denom, E2 + (size_t)p * HDD);

        // hypergraph branch on raw embeddings -> E1
        k_hg_t<<<EE, DD, 0, stream>>>(embeddings, cnt, codes, Dv, De, lens, p, T);
        k_hg_out<<<NN, DD, 0, stream>>>(cx, lens, p, T, Dv, zbuf);
        cvt(zbuf, tA, (size_t)NN * DD, stream);
        gemm(tA, WhBt, b_h, outB, NN, HDD, DD, 1, stream);
        k_pool<<<1, HDD, 0, stream>>>(outB, Dv, denom, E1 + (size_t)p * HDD);

        // hypergraph branch on icd_emb -> E3
        k_hg_t<<<EE, DD, 0, stream>>>(xbuf, cnt, codes, Dv, De, lens, p, T);
        k_hg_out<<<NN, DD, 0, stream>>>(cx, lens, p, T, Dv, zbuf);
        cvt(zbuf, tA, (size_t)NN * DD, stream);
        gemm(tA, WiBt, b_i, outB, NN, HDD, DD, 1, stream);
        k_pool<<<1, HDD, 0, stream>>>(outB, Dv, denom, E3 + (size_t)p * HDD);

        k_gate<<<1, HDD, 0, stream>>>(E1 + (size_t)p * HDD, E2 + (size_t)p * HDD,
                                      E3 + (size_t)p * HDD,
                                      gw_h, gb_h, gw_c, gb_c, gw_i, gb_i,
                                      resf + (size_t)p * HDD);
    }

    // ---- classifier (WMMA; M=16 fits in one 32-row tile group) ----
    cvt(resf, resB, (size_t)BB * HDD, stream);
    gemm(resB, WclsBt, b_cls, out, BB, OO, HDD, 0, stream);

    // ---- InfoNCE losses -> d_out[B*OUT] ----
    k_loss<<<1, 256, 0, stream>>>(E1, E2, E3, out + (size_t)BB * OO);
}
